// EfficientTransformerBlock_20229295964956
// MI455X (gfx1250) — matrix operand order repaired and checked by reading, not compiled, not run on hardware
//
#include <hip/hip_runtime.h>
#include <hip/hip_bf16.h>
#include <stdint.h>

// ---------------------------------------------------------------------------
// Transformer block on gfx1250: V_WMMA_F32_16X16X32_BF16 + TDM tile staging.
// B=4, T=2048, C=512, H=8, Dh=64, FF=1024.
// ---------------------------------------------------------------------------

#define B_  4
#define T_  2048
#define C_  512
#define H_  8
#define DH_ 64
#define FF_ 1024
#define ROWS_ (B_ * T_)          // 8192
#define QKVN_ (3 * C_)           // 1536
#define BK_   64                 // GEMM K-step staged per TDM transfer

typedef __bf16 v16bf __attribute__((ext_vector_type(16)));
typedef float  v8f   __attribute__((ext_vector_type(8)));
typedef unsigned int u32x4 __attribute__((ext_vector_type(4)));
typedef int          i32x4 __attribute__((ext_vector_type(4)));
typedef int          i32x8 __attribute__((ext_vector_type(8)));

union FragU { uint32_t u[8]; v16bf v; };

static __device__ __forceinline__ v8f wmma_bf16(v16bf a, v16bf b, v8f c) {
    return __builtin_amdgcn_wmma_f32_16x16x32_bf16(
        /*neg_a=*/false, a, /*neg_b=*/false, b,
        /*c_mod=*/(short)0, c, /*reuse_a=*/false, /*reuse_b=*/false);
}

// A-fragment (16x32 bf16, M x K): lane m = lane&15, half = lane>>4.
// VGPR j holds K pair at k = (j>>2)*16 + half*8 + (j&3)*2.
static __device__ __forceinline__ v16bf load_a_frag(const __bf16* A, int lda,
                                                    int m0, int k0, int lane) {
    FragU r;
    const int half = lane >> 4, m = lane & 15;
    const __bf16* base = A + (size_t)(m0 + m) * lda + k0 + (half << 3);
#pragma unroll
    for (int j = 0; j < 8; ++j) {
        const int k = ((j >> 2) << 4) + ((j & 3) << 1);
        r.u[j] = *reinterpret_cast<const uint32_t*>(base + k);
    }
    return r.v;
}

// B-fragment (32x16 bf16, K x N) from Bt stored [N][K] row-major:
// lane n = lane&15, half = lane>>4.  VGPR j holds K pair at k = (j>>2)*16 + half*8 + (j&3)*2.
static __device__ __forceinline__ v16bf load_b_frag(const __bf16* Bt, int ldbt,
                                                    int n0, int k0, int lane) {
    FragU r;
    const int half = lane >> 4, n = lane & 15;
    const __bf16* base = Bt + (size_t)(n0 + n) * ldbt + k0 + (half << 3);
#pragma unroll
    for (int j = 0; j < 8; ++j)
        r.u[j] = *reinterpret_cast<const uint32_t*>(base + ((j >> 2) << 4) + ((j & 3) << 1));
    return r.v;
}

// ---------------------------------------------------------------------------
// Tensor Data Mover: 2-D tile (tile_d0 elements fastest, tile_d1 rows) of a
// row-major bf16 tensor with row stride stride_d0 (elements) -> LDS.
// D# layout per CDNA5 ISA.
// ---------------------------------------------------------------------------
static __device__ __forceinline__ void tdm_load_2d_bf16(
    uint32_t lds_addr, const void* gptr,
    uint32_t tensor_d0, uint32_t tensor_d1,
    uint32_t tile_d0, uint32_t tile_d1, uint64_t stride_d0) {
    const uint64_t ga = (uint64_t)(uintptr_t)gptr;
    u32x4 g0;
    g0[0] = 1u;                                             // count=1, user mode
    g0[1] = lds_addr;                                       // LDS byte address
    g0[2] = (uint32_t)(ga & 0xFFFFFFFFu);                   // global_addr[31:0]
    g0[3] = ((uint32_t)(ga >> 32) & 0x01FFFFFFu) | 0x80000000u; // addr[56:32]|type=2
    i32x8 g1;
    g1[0] = (int)(1u << 16);                                // data_size=1 (2 bytes)
    g1[1] = (int)((tensor_d0 & 0xFFFFu) << 16);             // tensor_dim0[15:0]
    g1[2] = (int)(((tensor_d0 >> 16) & 0xFFFFu) |
                  ((tensor_d1 & 0xFFFFu) << 16));           // dim0 hi | dim1 lo
    g1[3] = (int)(((tensor_d1 >> 16) & 0xFFFFu) |
                  ((tile_d0 & 0xFFFFu) << 16));             // dim1 hi | tile_dim0
    g1[4] = (int)(tile_d1 & 0xFFFFu);                       // tile_dim1 (tile_dim2=0)
    g1[5] = (int)(uint32_t)(stride_d0 & 0xFFFFFFFFu);       // dim0_stride[31:0]
    g1[6] = (int)(uint32_t)((stride_d0 >> 32) & 0xFFFFu);   // dim0_stride[47:32]
    g1[7] = 0;                                              // dim1_stride unused (2D)
    const i32x4 z4 = (i32x4){0, 0, 0, 0};
#if __clang_major__ >= 23
    const i32x8 z8 = (i32x8){0, 0, 0, 0, 0, 0, 0, 0};
    __builtin_amdgcn_tensor_load_to_lds(g0, g1, z4, z4, z8, 0);
#else
    __builtin_amdgcn_tensor_load_to_lds(g0, g1, z4, z4, 0);
#endif
}

// ---------------------------------------------------------------------------
// Weight transpose-convert: W fp32 [K][N] -> Wt bf16 [N][K]
// ---------------------------------------------------------------------------
__global__ void wcvt_kernel(const float* __restrict__ W, __bf16* __restrict__ Wt,
                            int K, int N) {
    int idx = blockIdx.x * blockDim.x + threadIdx.x;
    if (idx < K * N) {
        int k = idx / N, n = idx - k * N;
        Wt[(size_t)n * K + k] = (__bf16)W[idx];
    }
}

// ---------------------------------------------------------------------------
// LayerNorm fp32 -> bf16, one 256-thread block per row (C = 512)
// ---------------------------------------------------------------------------
__global__ __launch_bounds__(256) void ln_kernel(const float* __restrict__ x,
                                                 const float* __restrict__ gamma,
                                                 const float* __restrict__ beta,
                                                 __bf16* __restrict__ out) {
    __shared__ float red[256];
    const int row = blockIdx.x, tid = threadIdx.x;
    const float* xr = x + (size_t)row * C_;
    float v0 = xr[tid], v1 = xr[tid + 256];

    red[tid] = v0 + v1;
    __syncthreads();
#pragma unroll
    for (int s = 128; s > 0; s >>= 1) {
        if (tid < s) red[tid] += red[tid + s];
        __syncthreads();
    }
    const float mu = red[0] * (1.0f / C_);
    __syncthreads();

    float d0 = v0 - mu, d1 = v1 - mu;
    red[tid] = d0 * d0 + d1 * d1;
    __syncthreads();
#pragma unroll
    for (int s = 128; s > 0; s >>= 1) {
        if (tid < s) red[tid] += red[tid + s];
        __syncthreads();
    }
    const float rstd = rsqrtf(red[0] * (1.0f / C_) + 1e-5f);

    out[(size_t)row * C_ + tid]       = (__bf16)(d0 * rstd * gamma[tid] + beta[tid]);
    out[(size_t)row * C_ + tid + 256] = (__bf16)(d1 * rstd * gamma[tid + 256] + beta[tid + 256]);
}

// ---------------------------------------------------------------------------
// TDM-staged WMMA GEMM:  out[M][N] = act(A[M][K] @ Bt[N][K]^T + bias) + resid
// Block = 256 threads (8 waves, 2x4) -> 128x128 tile; wave tile 64x32.
// Per K-step of 64: wave0 issues two TDM tile loads (A 128x64, B 128x64),
// waits TENSORcnt, barrier, then all waves read fragments from LDS.
// act: 0 = none, 1 = exact GELU.
// ---------------------------------------------------------------------------
__global__ __launch_bounds__(256) void gemm_kernel(const __bf16* __restrict__ A,
                                                   const __bf16* __restrict__ Bt,
                                                   const float* __restrict__ bias,
                                                   const float* __restrict__ resid,
                                                   float* __restrict__ outF,
                                                   __bf16* __restrict__ outB,
                                                   int M, int N, int K, int act) {
    __shared__ __bf16 ldsA[128 * BK_];   // [row 0..127][k 0..63]
    __shared__ __bf16 ldsB[128 * BK_];   // [n   0..127][k 0..63]

    const int lane = threadIdx.x & 31;
    const int wave = threadIdx.x >> 5;
    const int m0w = (wave & 1) * 64;     // wave tile inside block: 64 x 32
    const int n0w = (wave >> 1) * 32;
    const int bm = blockIdx.y * 128;
    const int bn = blockIdx.x * 128;

    const uint32_t ldsA_addr = (uint32_t)(uintptr_t)(&ldsA[0]);
    const uint32_t ldsB_addr = (uint32_t)(uintptr_t)(&ldsB[0]);

    v8f acc[4][2];
#pragma unroll
    for (int i = 0; i < 4; ++i)
#pragma unroll
        for (int j = 0; j < 2; ++j) acc[i][j] = (v8f){0, 0, 0, 0, 0, 0, 0, 0};

    for (int k0 = 0; k0 < K; k0 += BK_) {
        __syncthreads();                          // prior LDS consumption done
        if (threadIdx.x == 0) {                   // TDM ignores EXEC: one issue
            tdm_load_2d_bf16(ldsA_addr, A + (size_t)bm * K + k0,
                             (uint32_t)(K - k0), (uint32_t)(M - bm),
                             BK_, 128, (uint64_t)K);
            tdm_load_2d_bf16(ldsB_addr, Bt + (size_t)bn * K + k0,
                             (uint32_t)(K - k0), (uint32_t)(N - bn),
                             BK_, 128, (uint64_t)K);
            __builtin_amdgcn_s_wait_tensorcnt(0);
        }
        __syncthreads();                          // tiles visible to all waves

#pragma unroll
        for (int kk = 0; kk < BK_; kk += 32) {
            v16bf b0 = load_b_frag(ldsB, BK_, n0w,      kk, lane);
            v16bf b1 = load_b_frag(ldsB, BK_, n0w + 16, kk, lane);
#pragma unroll
            for (int i = 0; i < 4; ++i) {
                v16bf a = load_a_frag(ldsA, BK_, m0w + 16 * i, kk, lane);
                acc[i][0] = wmma_bf16(a, b0, acc[i][0]);
                acc[i][1] = wmma_bf16(a, b1, acc[i][1]);
            }
        }
    }

    const int half = lane >> 4, nn = lane & 15;
#pragma unroll
    for (int i = 0; i < 4; ++i)
#pragma unroll
        for (int j = 0; j < 2; ++j)
#pragma unroll
            for (int r = 0; r < 8; ++r) {
                const int m = bm + m0w + 16 * i + r + half * 8;
                const int n = bn + n0w + 16 * j + nn;
                float v = acc[i][j][r];
                if (bias) v += bias[n];
                if (act == 1) v = 0.5f * v * (1.0f + erff(v * 0.70710678118f));
                if (resid) v += resid[(size_t)m * N + n];
                const size_t o = (size_t)m * N + n;
                if (outF) outF[o] = v;
                if (outB) outB[o] = (__bf16)v;
            }
}

// ---------------------------------------------------------------------------
// V transpose scatter: qkv[b,t, 1024 + h*64 + d] -> vt[(b*H+h)*64 + d][t]
// ---------------------------------------------------------------------------
__global__ void vscat_kernel(const __bf16* __restrict__ qkv, __bf16* __restrict__ vt) {
    int idx = blockIdx.x * blockDim.x + threadIdx.x;
    if (idx >= B_ * T_ * H_ * DH_) return;
    int d = idx & 63, h = (idx >> 6) & 7, t = (idx >> 9) & 2047, b = idx >> 20;
    vt[(((size_t)(b * H_ + h)) * DH_ + d) * T_ + t] =
        qkv[((size_t)(b * T_ + t)) * QKVN_ + 2 * C_ + h * DH_ + d];
}

// ---------------------------------------------------------------------------
// Flash attention: one wave per (b, h, 16-query tile). Streams 32 keys/iter.
// Q/K read straight from the qkv buffer, V from vt. Output bf16 [ROWS][C].
// ---------------------------------------------------------------------------
__global__ __launch_bounds__(32) void attn_kernel(const __bf16* __restrict__ qkv,
                                                  const __bf16* __restrict__ vt,
                                                  __bf16* __restrict__ attnb) {
    __shared__ __bf16 lds_p[16 * 32];

    const int lane = threadIdx.x;
    const int half = lane >> 4, nn = lane & 15;
    const int tile = blockIdx.x;            // 0 .. B*H*(T/16)-1
    const int qt = tile & 127;
    const int h  = (tile >> 7) & 7;
    const int b  = tile >> 10;
    const int q0 = qt * 16;

    const __bf16* Aq = qkv + ((size_t)(b * T_ + q0)) * QKVN_ + h * DH_;          // Q rows
    const __bf16* Kt = qkv + ((size_t)(b * T_)) * QKVN_ + C_ + h * DH_;          // K as Bt (n=key)
    const __bf16* Vt = vt  + ((size_t)(b * H_ + h)) * DH_ * T_;                  // V as Bt (n=dh)

    v16bf aq0 = load_a_frag(Aq, QKVN_, 0, 0,  lane);
    v16bf aq1 = load_a_frag(Aq, QKVN_, 0, 32, lane);

    v8f o[4];
#pragma unroll
    for (int t = 0; t < 4; ++t) o[t] = (v8f){0, 0, 0, 0, 0, 0, 0, 0};
    float mrow[8], lsum[8];
#pragma unroll
    for (int r = 0; r < 8; ++r) { mrow[r] = -3.0e38f; lsum[r] = 0.0f; }

    for (int key0 = 0; key0 < T_; key0 += 32) {
        v8f s0 = (v8f){0, 0, 0, 0, 0, 0, 0, 0};
        v8f s1 = (v8f){0, 0, 0, 0, 0, 0, 0, 0};
        s0 = wmma_bf16(aq0, load_b_frag(Kt, QKVN_, key0,      0,  lane), s0);
        s0 = wmma_bf16(aq1, load_b_frag(Kt, QKVN_, key0,      32, lane), s0);
        s1 = wmma_bf16(aq0, load_b_frag(Kt, QKVN_, key0 + 16, 0,  lane), s1);
        s1 = wmma_bf16(aq1, load_b_frag(Kt, QKVN_, key0 + 16, 32, lane), s1);

        // online softmax; rows of the C-layout live across lanes 0-15 / 16-31
#pragma unroll
        for (int r = 0; r < 8; ++r) {
            float x0 = s0[r] * 0.125f;      // 1/sqrt(64)
            float x1 = s1[r] * 0.125f;
            float mx = fmaxf(x0, x1);
#pragma unroll
            for (int msk = 8; msk > 0; msk >>= 1) mx = fmaxf(mx, __shfl_xor(mx, msk, 16));
            const float mnew = fmaxf(mrow[r], mx);
            const float corr = __expf(mrow[r] - mnew);
            const float p0 = __expf(x0 - mnew);
            const float p1 = __expf(x1 - mnew);
            float sum = p0 + p1;
#pragma unroll
            for (int msk = 8; msk > 0; msk >>= 1) sum += __shfl_xor(sum, msk, 16);
            lsum[r] = lsum[r] * corr + sum;
            mrow[r] = mnew;
            o[0][r] *= corr; o[1][r] *= corr; o[2][r] *= corr; o[3][r] *= corr;
            lds_p[(r + half * 8) * 32 + nn]      = (__bf16)p0;
            lds_p[(r + half * 8) * 32 + 16 + nn] = (__bf16)p1;
        }
        __asm__ volatile("s_wait_dscnt 0" ::: "memory");  // LDS C->A layout bounce

        v16bf pa = load_a_frag(lds_p, 32, 0, 0, lane);
#pragma unroll
        for (int t = 0; t < 4; ++t)
            o[t] = wmma_bf16(pa, load_b_frag(Vt, T_, t * 16, key0, lane), o[t]);
    }

#pragma unroll
    for (int r = 0; r < 8; ++r) {
        const float inv = 1.0f / lsum[r];
        const size_t row = (size_t)(b * T_ + q0 + r + half * 8);
#pragma unroll
        for (int t = 0; t < 4; ++t)
            attnb[row * C_ + h * DH_ + t * 16 + nn] = (__bf16)(o[t][r] * inv);
    }
}

// ---------------------------------------------------------------------------
// Host-side orchestration
// ---------------------------------------------------------------------------
extern "C" void kernel_launch(void* const* d_in, const int* in_sizes, int n_in,
                              void* d_out, int out_size, void* d_ws, size_t ws_size,
                              hipStream_t stream) {
    const float* x      = (const float*)d_in[0];
    const float* w_qkv  = (const float*)d_in[1];
    const float* b_qkv  = (const float*)d_in[2];
    const float* w_proj = (const float*)d_in[3];
    const float* b_proj = (const float*)d_in[4];
    const float* w_ff1  = (const float*)d_in[5];
    const float* b_ff1  = (const float*)d_in[6];
    const float* w_ff2  = (const float*)d_in[7];
    const float* b_ff2  = (const float*)d_in[8];
    const float* g1     = (const float*)d_in[9];
    const float* be1    = (const float*)d_in[10];
    const float* g2     = (const float*)d_in[11];
    const float* be2    = (const float*)d_in[12];
    float* out          = (float*)d_out;

    uint8_t* p = (uint8_t*)d_ws;
    auto take = [&](size_t bytes) { uint8_t* q = p; p += (bytes + 255) & ~(size_t)255; return q; };

    __bf16* wqkvT = (__bf16*)take((size_t)QKVN_ * C_ * 2);   // [1536][512]
    __bf16* wprojT= (__bf16*)take((size_t)C_ * C_ * 2);      // [512][512]
    __bf16* wff1T = (__bf16*)take((size_t)FF_ * C_ * 2);     // [1024][512]
    __bf16* wff2T = (__bf16*)take((size_t)C_ * FF_ * 2);     // [512][1024]
    __bf16* h1    = (__bf16*)take((size_t)ROWS_ * C_ * 2);
    __bf16* qkv   = (__bf16*)take((size_t)ROWS_ * QKVN_ * 2);
    __bf16* vt    = (__bf16*)take((size_t)B_ * H_ * DH_ * T_ * 2);
    __bf16* attnb = (__bf16*)take((size_t)ROWS_ * C_ * 2);
    float*  y1    = (float*) take((size_t)ROWS_ * C_ * 4);
    __bf16* h2    = (__bf16*)take((size_t)ROWS_ * C_ * 2);
    __bf16* gbuf  = (__bf16*)take((size_t)ROWS_ * FF_ * 2);

    // 1) weights fp32 [K][N] -> bf16 [N][K]
    wcvt_kernel<<<(C_ * QKVN_ + 255) / 256, 256, 0, stream>>>(w_qkv,  wqkvT, C_,  QKVN_);
    wcvt_kernel<<<(C_ * C_    + 255) / 256, 256, 0, stream>>>(w_proj, wprojT, C_, C_);
    wcvt_kernel<<<(C_ * FF_   + 255) / 256, 256, 0, stream>>>(w_ff1,  wff1T, C_,  FF_);
    wcvt_kernel<<<(FF_ * C_   + 255) / 256, 256, 0, stream>>>(w_ff2,  wff2T, FF_, C_);

    // 2) LN1
    ln_kernel<<<ROWS_, 256, 0, stream>>>(x, g1, be1, h1);

    // 3) QKV GEMM (bias fused), bf16 out
    gemm_kernel<<<dim3(QKVN_ / 128, ROWS_ / 128), 256, 0, stream>>>(
        h1, wqkvT, b_qkv, nullptr, nullptr, qkv, ROWS_, QKVN_, C_, 0);

    // 4) V transpose scatter
    vscat_kernel<<<(B_ * T_ * H_ * DH_ + 255) / 256, 256, 0, stream>>>(qkv, vt);

    // 5) flash attention (one wave per 16-query tile)
    attn_kernel<<<B_ * H_ * (T_ / 16), 32, 0, stream>>>(qkv, vt, attnb);

    // 6) proj GEMM + bias + residual(x) -> y1 fp32
    gemm_kernel<<<dim3(C_ / 128, ROWS_ / 128), 256, 0, stream>>>(
        attnb, wprojT, b_proj, x, y1, nullptr, ROWS_, C_, C_, 0);

    // 7) LN2
    ln_kernel<<<ROWS_, 256, 0, stream>>>(y1, g2, be2, h2);

    // 8) FF1 GEMM + bias + GELU -> bf16
    gemm_kernel<<<dim3(FF_ / 128, ROWS_ / 128), 256, 0, stream>>>(
        h2, wff1T, b_ff1, nullptr, nullptr, gbuf, ROWS_, FF_, C_, 1);

    // 9) FF2 GEMM + bias + residual(y1) -> d_out fp32
    gemm_kernel<<<dim3(C_ / 128, ROWS_ / 128), 256, 0, stream>>>(
        gbuf, wff2T, b_ff2, y1, out, nullptr, ROWS_, C_, FF_, 0);

    (void)in_sizes; (void)n_in; (void)out_size; (void)ws_size;
}
